// HJiT_2920577761570
// MI455X (gfx1250) — compile-verified, matrix operand order checked by reading, not run on hardware
//
#include <hip/hip_runtime.h>
#include <hip/hip_bf16.h>
#include <math.h>

// ---------------------------------------------------------------------------
// Types / constants
// ---------------------------------------------------------------------------
typedef __attribute__((ext_vector_type(16))) _Float16 v16h;
typedef __attribute__((ext_vector_type(8)))  float    v8f;
typedef __attribute__((ext_vector_type(4)))  unsigned u32x4;
typedef __attribute__((ext_vector_type(8)))  int      i32x8;
typedef __attribute__((ext_vector_type(4)))  int      i32x4;

constexpr int NB    = 4;     // batch
constexpr int NCOND = 1024;  // conditioning dim
constexpr int NHEAD = 16;

struct Blk {
    const float *norm1, *norm2, *qkv_w, *qkv_b, *q_norm, *k_norm,
                *proj_w, *proj_b, *w12_w, *w12_b, *w3_w, *w3_b,
                *ada_w, *ada_b;
};

// ---------------------------------------------------------------------------
// LDS byte-offset of a shared-memory pointer (generic -> addrspace(3) -> int)
// ---------------------------------------------------------------------------
typedef __attribute__((address_space(3))) char lds_char;
__device__ inline unsigned lds_offset(const void* p)
{
    return (unsigned)(unsigned long long)(lds_char*)p;
}

// ---------------------------------------------------------------------------
// Tensor Data Mover: async 2D tile load (f16 elements) global -> LDS.
// D# per CDNA5 ISA:
//   group0: count=1 | lds_addr | global_addr | type=2
//   group1: data_size=1(2B), tensor_dim0/1 = remaining cols/rows (OOB -> 0),
//           tile_dim0/1, tensor_dim0_stride = row stride (elements)
// Issued by one wave; completion tracked with TENSORcnt.
// ---------------------------------------------------------------------------
__device__ inline void tdm_load_2d_f16(unsigned lds_byte, const _Float16* gptr,
                                       int rows_rem, int cols_rem,
                                       int row_stride_elems,
                                       int tile_rows, int tile_cols)
{
    unsigned long long ga = (unsigned long long)gptr;
    u32x4 g0;
    g0[0] = 1u;                                   // count = 1 (valid descriptor)
    g0[1] = lds_byte;                             // LDS destination (bytes)
    g0[2] = (unsigned)(ga & 0xFFFFFFFFull);       // global_addr[31:0]
    g0[3] = (unsigned)((ga >> 32) & 0x01FFFFFFull) | (2u << 30);  // [56:32]|type=2

    i32x8 g1;
    g1[0] = 0x00010000;                           // data_size = 1 -> 2 bytes
    g1[1] = (int)((unsigned)(cols_rem & 0xFFFF) << 16);            // tdim0 lo
    g1[2] = (int)((((unsigned)cols_rem >> 16) & 0xFFFFu)
                  | ((unsigned)(rows_rem & 0xFFFF) << 16));        // tdim0 hi|tdim1 lo
    g1[3] = (int)(((unsigned)rows_rem >> 16) & 0xFFFFu)
            | (tile_cols << 16);                  // tdim1 hi | tile_dim0
    g1[4] = tile_rows & 0xFFFF;                   // tile_dim1 (tile_dim2 = 0)
    g1[5] = row_stride_elems;                     // tensor_dim0_stride lo32
    g1[6] = 0;
    g1[7] = 0;

    // Force descriptor dwords into SGPRs (D# lives in SGPRs per ISA).
#pragma unroll
    for (int i = 0; i < 4; ++i)
        g0[i] = (unsigned)__builtin_amdgcn_readfirstlane((int)g0[i]);
#pragma unroll
    for (int i = 0; i < 8; ++i)
        g1[i] = __builtin_amdgcn_readfirstlane(g1[i]);

    i32x4 z4 = {0, 0, 0, 0};
#if defined(__clang_major__) && (__clang_major__ >= 23)
    i32x8 z8 = {0, 0, 0, 0, 0, 0, 0, 0};
    __builtin_amdgcn_tensor_load_to_lds(g0, g1, z4, z4, z8, 0);
#else
    __builtin_amdgcn_tensor_load_to_lds(g0, g1, z4, z4, 0);
#endif
}

// ---------------------------------------------------------------------------
// WMMA GEMM:  D[M,N](f32) = act( A[M,K](f16) * W[N,K](f16)^T + bias[N](f32) )
// 256 threads = 8 waves; tile 128(M) x 64(N); K staged 32 wide via TDM with
// DOUBLE BUFFERING: wave 0 issues tile i+1 while all waves compute tile i.
// TDM ops complete in order per wave, so s_wait_tensorcnt <=2 guarantees the
// two descriptors of tile i are done while tile i+1's two are still in flight.
// Wave w computes rows [16w,16w+16) of the tile with 4x v_wmma_f32_16x16x32_f16.
// ---------------------------------------------------------------------------
#define GT_M 128
#define GT_N 64
#define GT_K 32

__global__ __launch_bounds__(256) void k_gemm(
    const _Float16* __restrict__ A, const _Float16* __restrict__ W,
    const float* __restrict__ bias, float* __restrict__ D,
    int M, int N, int K, int act)
{
    __shared__ alignas(64) _Float16 sA[2][GT_M * GT_K];   // 2 x 8 KB
    __shared__ alignas(64) _Float16 sB[2][GT_N * GT_K];   // 2 x 4 KB

    const int tid  = threadIdx.x;
    const int wave = tid >> 5;
    const int lane = tid & 31;
    const int m0   = blockIdx.y * GT_M;
    const int n0   = blockIdx.x * GT_N;

    unsigned ldsA[2] = { lds_offset(sA[0]), lds_offset(sA[1]) };
    unsigned ldsB[2] = { lds_offset(sB[0]), lds_offset(sB[1]) };

    const _Float16* Ab = A + (size_t)m0 * K;
    const _Float16* Wb = W + (size_t)n0 * K;

    v8f acc[4] = {};

    const int nk = (K + GT_K - 1) / GT_K;

    // Preload tile 0 into buffer 0.
    if (wave == 0) {
        tdm_load_2d_f16(ldsA[0], Ab, M - m0, K, K, GT_M, GT_K);
        tdm_load_2d_f16(ldsB[0], Wb, N - n0, K, K, GT_N, GT_K);
    }

    for (int i = 0; i < nk; ++i) {
        const int cur = i & 1;
        if (wave == 0) {
            if (i + 1 < nk) {
                int kt = (i + 1) * GT_K;   // prefetch next K-tile (overlapped)
                tdm_load_2d_f16(ldsA[cur ^ 1], Ab + kt, M - m0, K - kt, K,
                                GT_M, GT_K);
                tdm_load_2d_f16(ldsB[cur ^ 1], Wb + kt, N - n0, K - kt, K,
                                GT_N, GT_K);
                __builtin_amdgcn_s_wait_tensorcnt(2);   // tile i complete
            } else {
                __builtin_amdgcn_s_wait_tensorcnt(0);   // last tile complete
            }
        }
        __syncthreads();                   // tile i visible to all waves

        const _Float16* pa = sA[cur];
        const _Float16* pb = sB[cur];
        const int row = wave * 16 + (lane & 15);
        const int kh  = (lane >> 4) * 16;
        v16h afrag = *(const v16h*)&pa[row * GT_K + kh];
#pragma unroll
        for (int t = 0; t < 4; ++t) {
            int bn = t * 16 + (lane & 15);
            v16h bfrag = *(const v16h*)&pb[bn * GT_K + kh];
            acc[t] = __builtin_amdgcn_wmma_f32_16x16x32_f16(
                false, afrag, false, bfrag, (short)0, acc[t], false, false);
        }
        __syncthreads();                   // tile i consumed -> buffer reusable
    }

    // Epilogue: D layout -> n = lane%16, m = j + 8*(lane/16) inside 16x16 tile
#pragma unroll
    for (int t = 0; t < 4; ++t) {
        int n = n0 + t * 16 + (lane & 15);
        if (n >= N) continue;
        float bv = bias ? bias[n] : 0.f;
#pragma unroll
        for (int j = 0; j < 8; ++j) {
            int m = m0 + wave * 16 + j + 8 * (lane >> 4);
            if (m >= M) continue;
            float v = acc[t][j] + bv;
            if (act == 1) v = v / (1.f + expf(-v));   // SiLU
            D[(size_t)m * N + n] = v;
        }
    }
}

// ---------------------------------------------------------------------------
// Elementwise / small kernels
// ---------------------------------------------------------------------------
__global__ void k_cvt(const float* __restrict__ x, _Float16* __restrict__ y,
                      int total)
{
    int i = blockIdx.x * 256 + threadIdx.x;
    if (i >= total) return;
    y[i] = (_Float16)x[i];
}

__global__ void k_timestep(const float* __restrict__ t, _Float16* __restrict__ temb)
{
    int b = blockIdx.x, i = threadIdx.x;                 // 256 threads
    float fr  = expf(-logf(10000.f) * (float)(i & 127) / 128.f);
    float arg = t[b] * fr;
    temb[b * 256 + i] = (_Float16)((i < 128) ? cosf(arg) : sinf(arg));
}

__global__ void k_silu(const float* __restrict__ x, _Float16* __restrict__ y,
                       int total)
{
    int i = blockIdx.x * 256 + threadIdx.x;
    if (i >= total) return;
    float v = x[i];
    y[i] = (_Float16)(v / (1.f + expf(-v)));
}

// EVA-style axial 2D RoPE tables: cos/sin [N, hd]
__global__ void k_rope_init(float* __restrict__ cosT, float* __restrict__ sinT,
                            int side, int hd)
{
    int n = blockIdx.x, j = threadIdx.x;
    if (j >= hd) return;
    int da = hd >> 1;
    int pos, jj;
    if (j < da) { pos = n / side; jj = j; }          // height axis
    else        { pos = n % side; jj = j - da; }     // width axis
    float ex   = (float)(2 * (jj >> 1)) / (float)da;
    float freq = powf(10000.f, -ex);
    float ang  = (float)pos * freq;
    cosT[n * hd + j] = cosf(ang);
    sinT[n * hd + j] = sinf(ang);
}

// y(f16) = rmsnorm(x)*w * (1+scale) + shift ; shift/scale per-(b,c) mod slices
__global__ __launch_bounds__(256) void k_rms_mod(
    const float* __restrict__ x, const float* __restrict__ w,
    const float* __restrict__ mod, int modStride, int shiftOff, int scaleOff,
    _Float16* __restrict__ y, int Ntok, int C)
{
    __shared__ float red[256];
    int tok = blockIdx.x;                 // b*Ntok + n
    int b   = tok / Ntok;
    const float* xp = x + (size_t)tok * C;
    float ss = 0.f;
    for (int c = threadIdx.x; c < C; c += 256) { float v = xp[c]; ss += v * v; }
    red[threadIdx.x] = ss;
    __syncthreads();
    for (int st = 128; st > 0; st >>= 1) {
        if (threadIdx.x < st) red[threadIdx.x] += red[threadIdx.x + st];
        __syncthreads();
    }
    float r = rsqrtf(red[0] / (float)C + 1e-6f);
    const float* mb = mod + (size_t)b * modStride;
    _Float16* yp = y + (size_t)tok * C;
    for (int c = threadIdx.x; c < C; c += 256)
        yp[c] = (_Float16)(xp[c] * r * w[c] * (1.f + mb[scaleOff + c])
                           + mb[shiftOff + c]);
}

// qkv[B*N, 3C](f32) -> q/k/v [B,H,N,hd](f32), per-head RMSNorm(q,k) + RoPE(q,k)
__global__ void k_qkv_post(
    const float* __restrict__ qkv, const float* __restrict__ qn,
    const float* __restrict__ kn, const float* __restrict__ cosT,
    const float* __restrict__ sinT, float* __restrict__ q,
    float* __restrict__ k, float* __restrict__ v, int Ntok, int H, int hd)
{
    int tok = blockIdx.x;                 // b*Ntok + n
    int h   = threadIdx.x;
    if (h >= H) return;
    int n = tok % Ntok, b = tok / Ntok;
    int C = H * hd;
    const float* base = qkv + (size_t)tok * 3 * C;
    const float* qp = base + h * hd;
    const float* kp = base + C + h * hd;
    const float* vp = base + 2 * C + h * hd;
    size_t obase = (((size_t)b * H + h) * Ntok + n) * hd;
    const float* ct = cosT + (size_t)n * hd;
    const float* st = sinT + (size_t)n * hd;

    float ssq = 0.f, ssk = 0.f;
    for (int d = 0; d < hd; ++d) { ssq += qp[d] * qp[d]; ssk += kp[d] * kp[d]; }
    float rq = rsqrtf(ssq / (float)hd + 1e-6f);
    float rk = rsqrtf(ssk / (float)hd + 1e-6f);

    for (int d = 0; d < hd; d += 2) {
        float c0 = ct[d], s0 = st[d], c1 = ct[d + 1], s1 = st[d + 1];
        float a = qp[d] * rq * qn[d], bb = qp[d + 1] * rq * qn[d + 1];
        q[obase + d]     = a * c0 - bb * s0;      // x*cos + rot_half(x)*sin
        q[obase + d + 1] = bb * c1 + a * s1;
        float e = kp[d] * rk * kn[d], f = kp[d + 1] * rk * kn[d + 1];
        k[obase + d]     = e * c0 - f * s0;
        k[obase + d + 1] = f * c1 + e * s1;
    }
    for (int d = 0; d < hd; ++d) v[obase + d] = vp[d];
}

// softmax attention for one (b,h,query); scores in LDS (N<=1024).
// Output (f16) written in [B, N, H*hd] layout for the proj GEMM.
__global__ __launch_bounds__(256) void k_attn(
    const float* __restrict__ q, const float* __restrict__ k,
    const float* __restrict__ v, _Float16* __restrict__ o,
    int Ntok, int H, int hd, float scale)
{
    __shared__ float s_sc[1024];
    __shared__ float s_q[64];
    __shared__ float red[256];
    int nq = blockIdx.x, bh = blockIdx.y;
    int b = bh / H, h = bh % H;
    int t = threadIdx.x;
    const float* qp = q + ((size_t)bh * Ntok + nq) * hd;
    for (int d = t; d < hd; d += 256) s_q[d] = qp[d];
    __syncthreads();

    float lmax = -1e30f;
    for (int m = t; m < Ntok; m += 256) {
        const float* kp = k + ((size_t)bh * Ntok + m) * hd;
        float s = 0.f;
        for (int d = 0; d < hd; ++d) s += s_q[d] * kp[d];
        s *= scale;
        s_sc[m] = s;
        lmax = fmaxf(lmax, s);
    }
    red[t] = lmax; __syncthreads();
    for (int st = 128; st > 0; st >>= 1) {
        if (t < st) red[t] = fmaxf(red[t], red[t + st]);
        __syncthreads();
    }
    float mx = red[0];
    __syncthreads();

    float lsum = 0.f;
    for (int m = t; m < Ntok; m += 256) {
        float e = expf(s_sc[m] - mx);
        s_sc[m] = e; lsum += e;
    }
    red[t] = lsum; __syncthreads();
    for (int st = 128; st > 0; st >>= 1) {
        if (t < st) red[t] += red[t + st];
        __syncthreads();
    }
    float inv = 1.f / red[0];
    __syncthreads();

    for (int d = t; d < hd; d += 256) {
        float s = 0.f;
        for (int m = 0; m < Ntok; ++m)
            s += s_sc[m] * v[((size_t)bh * Ntok + m) * hd + d];
        o[(((size_t)b * Ntok + nq) * H + h) * hd + d] = (_Float16)(s * inv);
    }
}

// x += gate[b,c] * y   (gate = slice of mod)
__global__ void k_gated_add(float* __restrict__ x, const float* __restrict__ y,
                            const float* __restrict__ mod, int modStride,
                            int gOff, int Ntok, int C, int total)
{
    int i = blockIdx.x * 256 + threadIdx.x;
    if (i >= total) return;
    int c = i % C;
    int b = i / (Ntok * C);
    x[i] += mod[(size_t)b * modStride + gOff + c] * y[i];
}

// out(f16)[row, j] = silu(h12[row, j]) * h12[row, mh + j]
__global__ void k_swiglu(const float* __restrict__ h12, _Float16* __restrict__ out,
                         int mh, int total)
{
    int i = blockIdx.x * 256 + threadIdx.x;
    if (i >= total) return;
    int row = i / mh, j = i % mh;
    float a = h12[(size_t)row * 2 * mh + j];
    float g = h12[(size_t)row * 2 * mh + mh + j];
    out[i] = (_Float16)((a / (1.f + expf(-a))) * g);
}

// pixel-shuffle merge: [B, Hside^2, Cin](f32) -> [B, (Hside/ph)^2, ph*ph*Cin](f16)
__global__ void k_merge(const float* __restrict__ x, _Float16* __restrict__ y,
                        int Hside, int ph, int Cin, int total)
{
    int idx = blockIdx.x * 256 + threadIdx.x;
    if (idx >= total) return;
    int Co = ph * ph * Cin;
    int Ho = Hside / ph;
    int No = Ho * Ho;
    int qd = idx % Co;
    int p  = (idx / Co) % No;
    int b  = idx / (Co * No);
    int i  = qd / (ph * Cin);
    int rem = qd % (ph * Cin);
    int j  = rem / Cin, c = rem % Cin;
    int h2 = p / Ho, w2 = p % Ho;
    size_t src = ((size_t)b * Hside * Hside + (size_t)(h2 * ph + i) * Hside
                  + (w2 * ph + j)) * Cin + c;
    y[idx] = (_Float16)x[src];
}

// inverse pixel-shuffle + lerp vs skip:
// y [B, H^2, ph*ph*Cout] -> out [B, (H*ph)^2, Cout]; out = skip + fac*(y - skip)
__global__ void k_split(const float* __restrict__ y, const float* __restrict__ skip,
                        const float* __restrict__ fac, float* __restrict__ out,
                        int H, int ph, int Cout, int total)
{
    int idx = blockIdx.x * 256 + threadIdx.x;
    if (idx >= total) return;
    int W2 = H * ph;
    int No = W2 * W2;
    int c  = idx % Cout;
    int p  = (idx / Cout) % No;
    int b  = idx / (Cout * No);
    int orow = p / W2, ocol = p % W2;
    int h = orow / ph, i = orow % ph;
    int w = ocol / ph, j = ocol % ph;
    float yv = y[(((size_t)b * H * H + (size_t)h * H + w) * (ph * ph * Cout))
                 + (size_t)(i * ph + j) * Cout + c];
    float sk = skip[idx];
    out[idx] = sk + fac[0] * (yv - sk);
}

__global__ void k_add_pos(float* __restrict__ x, const float* __restrict__ pos,
                          int perBatch, int total)
{
    int i = blockIdx.x * 256 + threadIdx.x;
    if (i >= total) return;
    x[i] += pos[i % perBatch];
}

// ---------------------------------------------------------------------------
// Host orchestration
// ---------------------------------------------------------------------------
extern "C" void kernel_launch(void* const* d_in, const int* in_sizes, int n_in,
                              void* d_out, int out_size, void* d_ws, size_t ws_size,
                              hipStream_t stream)
{
    (void)in_sizes; (void)n_in; (void)out_size; (void)ws_size;

    const float* x_in = (const float*)d_in[0];
    const float* t_in = (const float*)d_in[1];
    int pi = 2;
    auto nxt = [&]() -> const float* { return (const float*)d_in[pi++]; };

    const float* t_w1 = nxt(); const float* t_b1 = nxt();
    const float* t_w2 = nxt(); const float* t_b2 = nxt();
    const float* x_emb_w   = nxt();
    const float* pos_embed = nxt();

    auto readBlk = [&]() {
        Blk p;
        p.norm1 = nxt(); p.norm2 = nxt();
        p.qkv_w = nxt(); p.qkv_b = nxt();
        p.q_norm = nxt(); p.k_norm = nxt();
        p.proj_w = nxt(); p.proj_b = nxt();
        p.w12_w = nxt(); p.w12_b = nxt();
        p.w3_w  = nxt(); p.w3_b  = nxt();
        p.ada_w = nxt(); p.ada_b = nxt();
        return p;
    };
    Blk down0[2], down1[2], down2[4], midb[2], up0[4], up1[2], up2[2];
    for (int j = 0; j < 2; ++j) down0[j] = readBlk();
    for (int j = 0; j < 2; ++j) down1[j] = readBlk();
    for (int j = 0; j < 4; ++j) down2[j] = readBlk();
    for (int j = 0; j < 2; ++j) midb[j]  = readBlk();
    for (int j = 0; j < 4; ++j) up0[j]   = readBlk();   // C=512
    for (int j = 0; j < 2; ++j) up1[j]   = readBlk();   // C=256
    for (int j = 0; j < 2; ++j) up2[j]   = readBlk();   // C=128
    const float* merge_w[3];   for (int i = 0; i < 3; ++i) merge_w[i]   = nxt();
    const float* split_w[3];   for (int i = 0; i < 3; ++i) split_w[i]   = nxt();
    const float* split_fac[3]; for (int i = 0; i < 3; ++i) split_fac[i] = nxt();
    const float* final_norm  = nxt();
    const float* final_w     = nxt();
    const float* final_b     = nxt();
    const float* final_ada_w = nxt();
    const float* final_ada_b = nxt();

    // ---- workspace carve-up (units: floats) ----
    float* ws = (float*)d_ws;
    size_t off = 0;
    auto alloc = [&](size_t n) { float* p = ws + off; off += n; return p; };
    _Float16* temb_h  = (_Float16*)alloc(512);              // B*256 halves
    float*    c1f     = alloc((size_t)NB * NCOND);
    _Float16* c1h     = (_Float16*)alloc(2048);             // B*1024 halves
    float*    cbuf    = alloc((size_t)NB * NCOND);
    _Float16* csilu_h = (_Float16*)alloc(2048);             // B*1024 halves
    float*    modb    = alloc((size_t)NB * 6 * 1024);
    float*    mod2    = alloc((size_t)NB * 256);
    float*    xa      = alloc(524288);                      // residual stream
    float*    big     = alloc(2793472);                     // qkv / mlp-h12 (f32)
    float*    qb      = alloc(524288);
    float*    kb      = alloc(524288);
    float*    vb      = alloc(524288);
    float*    yb      = alloc(524288);
    _Float16* ah      = (_Float16*)alloc(698368);           // 1,396,736 halves: GEMM A
    _Float16* wh      = (_Float16*)alloc(3145728);          // 6,291,456 halves: weights
    float*    skip0   = alloc(524288);
    float*    skip1   = alloc(262144);
    float*    skip2   = alloc(131072);
    int sides[4] = {32, 16, 8, 4};
    int hds[4]   = {8, 16, 32, 64};
    float *cosT[4], *sinT[4];
    for (int l = 0; l < 4; ++l) {
        int n = sides[l] * sides[l] * hds[l];
        cosT[l] = alloc(n); sinT[l] = alloc(n);
    }

    auto cvtTo = [&](const float* src, _Float16* dst, int n) {
        k_cvt<<<(n + 255) / 256, 256, 0, stream>>>(src, dst, n);
    };
    auto cvtW = [&](const float* src, int n) { cvtTo(src, wh, n); };
    auto gemm = [&](const _Float16* A, const float* bias, float* Dd,
                    int M, int N, int K, int act) {
        dim3 g((unsigned)((N + GT_N - 1) / GT_N), (unsigned)((M + GT_M - 1) / GT_M));
        k_gemm<<<g, 256, 0, stream>>>(A, wh, bias, Dd, M, N, K, act);
    };

    auto run_block = [&](const Blk& p, int lvl, int Ntok, int C) {
        int hd = C / NHEAD;
        int mh = (4 * C * 2) / 3;                      // SwiGLU hidden
        // adaLN modulation: [B, 6C]
        cvtW(p.ada_w, 6 * C * NCOND);
        gemm(csilu_h, p.ada_b, modb, NB, 6 * C, NCOND, 0);
        // attn branch
        k_rms_mod<<<NB * Ntok, 256, 0, stream>>>(xa, p.norm1, modb, 6 * C, 0, C,
                                                 ah, Ntok, C);
        cvtW(p.qkv_w, 3 * C * C);
        gemm(ah, p.qkv_b, big, NB * Ntok, 3 * C, C, 0);
        k_qkv_post<<<NB * Ntok, 32, 0, stream>>>(big, p.q_norm, p.k_norm,
                                                 cosT[lvl], sinT[lvl],
                                                 qb, kb, vb, Ntok, NHEAD, hd);
        k_attn<<<dim3(Ntok, NB * NHEAD), 256, 0, stream>>>(
            qb, kb, vb, ah, Ntok, NHEAD, hd, 1.0f / sqrtf((float)hd));
        cvtW(p.proj_w, C * C);
        gemm(ah, p.proj_b, yb, NB * Ntok, C, C, 0);
        { int tot = NB * Ntok * C;
          k_gated_add<<<(tot + 255) / 256, 256, 0, stream>>>(
              xa, yb, modb, 6 * C, 2 * C, Ntok, C, tot); }
        // mlp branch
        k_rms_mod<<<NB * Ntok, 256, 0, stream>>>(xa, p.norm2, modb, 6 * C,
                                                 3 * C, 4 * C, ah, Ntok, C);
        cvtW(p.w12_w, 2 * mh * C);
        gemm(ah, p.w12_b, big, NB * Ntok, 2 * mh, C, 0);
        { int tot = NB * Ntok * mh;
          k_swiglu<<<(tot + 255) / 256, 256, 0, stream>>>(big, ah, mh, tot); }
        cvtW(p.w3_w, C * mh);
        gemm(ah, p.w3_b, yb, NB * Ntok, C, mh, 0);
        { int tot = NB * Ntok * C;
          k_gated_add<<<(tot + 255) / 256, 256, 0, stream>>>(
              xa, yb, modb, 6 * C, 5 * C, Ntok, C, tot); }
    };

    // ---- RoPE tables ----
    for (int l = 0; l < 4; ++l)
        k_rope_init<<<sides[l] * sides[l], 64, 0, stream>>>(cosT[l], sinT[l],
                                                            sides[l], hds[l]);

    // ---- conditioning: t -> temb -> c -> silu(c) ----
    k_timestep<<<NB, 256, 0, stream>>>(t_in, temb_h);
    cvtW(t_w1, NCOND * 256);
    gemm(temb_h, t_b1, c1f, NB, NCOND, 256, 1);          // fused SiLU
    cvtTo(c1f, c1h, NB * NCOND);
    cvtW(t_w2, NCOND * NCOND);
    gemm(c1h, t_b2, cbuf, NB, NCOND, NCOND, 0);
    { int tot = NB * NCOND;
      k_silu<<<(tot + 255) / 256, 256, 0, stream>>>(cbuf, csilu_h, tot); }

    // ---- patchify + embed + pos ----
    { int tot = NB * 256 * 256 * 3;
      k_merge<<<(tot + 255) / 256, 256, 0, stream>>>(x_in, ah, 256, 8, 3, tot); }
    cvtW(x_emb_w, 128 * 192);
    gemm(ah, nullptr, xa, NB * 1024, 128, 192, 0);
    { int tot = NB * 1024 * 128;
      k_add_pos<<<(tot + 255) / 256, 256, 0, stream>>>(xa, pos_embed,
                                                       1024 * 128, tot); }

    // ---- down path ----
    for (int j = 0; j < 2; ++j) run_block(down0[j], 0, 1024, 128);
    hipMemcpyAsync(skip0, xa, (size_t)NB * 1024 * 128 * sizeof(float),
                   hipMemcpyDeviceToDevice, stream);
    { int tot = NB * 32 * 32 * 128;
      k_merge<<<(tot + 255) / 256, 256, 0, stream>>>(xa, ah, 32, 2, 128, tot); }
    cvtW(merge_w[0], 256 * 512);
    gemm(ah, nullptr, xa, NB * 256, 256, 512, 0);

    for (int j = 0; j < 2; ++j) run_block(down1[j], 1, 256, 256);
    hipMemcpyAsync(skip1, xa, (size_t)NB * 256 * 256 * sizeof(float),
                   hipMemcpyDeviceToDevice, stream);
    { int tot = NB * 16 * 16 * 256;
      k_merge<<<(tot + 255) / 256, 256, 0, stream>>>(xa, ah, 16, 2, 256, tot); }
    cvtW(merge_w[1], 512 * 1024);
    gemm(ah, nullptr, xa, NB * 64, 512, 1024, 0);

    for (int j = 0; j < 4; ++j) run_block(down2[j], 2, 64, 512);
    hipMemcpyAsync(skip2, xa, (size_t)NB * 64 * 512 * sizeof(float),
                   hipMemcpyDeviceToDevice, stream);
    { int tot = NB * 8 * 8 * 512;
      k_merge<<<(tot + 255) / 256, 256, 0, stream>>>(xa, ah, 8, 2, 512, tot); }
    cvtW(merge_w[2], 1024 * 2048);
    gemm(ah, nullptr, xa, NB * 16, 1024, 2048, 0);

    // ---- mid ----
    for (int j = 0; j < 2; ++j) run_block(midb[j], 3, 16, 1024);

    // ---- up path (split = proj + inverse shuffle + lerp vs skip) ----
    cvtTo(xa, ah, NB * 16 * 1024);
    cvtW(split_w[0], 2048 * 1024);
    gemm(ah, nullptr, yb, NB * 16, 2048, 1024, 0);
    { int tot = NB * 64 * 512;
      k_split<<<(tot + 255) / 256, 256, 0, stream>>>(yb, skip2, split_fac[0],
                                                     xa, 4, 2, 512, tot); }
    for (int j = 0; j < 4; ++j) run_block(up0[j], 2, 64, 512);

    cvtTo(xa, ah, NB * 64 * 512);
    cvtW(split_w[1], 1024 * 512);
    gemm(ah, nullptr, yb, NB * 64, 1024, 512, 0);
    { int tot = NB * 256 * 256;
      k_split<<<(tot + 255) / 256, 256, 0, stream>>>(yb, skip1, split_fac[1],
                                                     xa, 8, 2, 256, tot); }
    for (int j = 0; j < 2; ++j) run_block(up1[j], 1, 256, 256);

    cvtTo(xa, ah, NB * 256 * 256);
    cvtW(split_w[2], 512 * 256);
    gemm(ah, nullptr, yb, NB * 256, 512, 256, 0);
    { int tot = NB * 1024 * 128;
      k_split<<<(tot + 255) / 256, 256, 0, stream>>>(yb, skip0, split_fac[2],
                                                     xa, 16, 2, 128, tot); }
    for (int j = 0; j < 2; ++j) run_block(up2[j], 0, 1024, 128);

    // ---- final layer ----
    cvtW(final_ada_w, 256 * NCOND);
    gemm(csilu_h, final_ada_b, mod2, NB, 256, NCOND, 0);
    k_rms_mod<<<NB * 1024, 256, 0, stream>>>(xa, final_norm, mod2, 256, 0, 128,
                                             ah, 1024, 128);
    cvtW(final_w, 192 * 128);
    gemm(ah, final_b, (float*)d_out, NB * 1024, 192, 128, 0);
}
